// LSTMGCNModel_83021717832866
// MI455X (gfx1250) — matrix operand order repaired and checked by reading, compiled, not run on hardware
//
#include <hip/hip_runtime.h>
#include <hip/hip_bf16.h>

// CDNA5 / gfx1250: wave32, WMMA f32_16x16x32_f16.
// Only the temporal (LSTM) branch of the reference is live; GCN branch is dead.

typedef __attribute__((ext_vector_type(16))) _Float16 v16h;
typedef __attribute__((ext_vector_type(8)))  _Float16 v8h;
typedef __attribute__((ext_vector_type(8)))  float    v8f;

#define NWAVES 8
#define TSTEPS 12
#define PITCH  40   // halfs per h-row: 32 data + 8 pad (keeps 16B alignment: 80B rows)

__device__ __forceinline__ float sigm_f(float x) {
    return 1.0f / (1.0f + __expf(-x));
}
__device__ __forceinline__ float tanh_f(float x) {
    float xc = fminf(fmaxf(x, -15.0f), 15.0f);
    float e  = __expf(-2.0f * xc);
    return (1.0f - e) / (1.0f + e);
}

#define WMMA16(a, b, c) \
    __builtin_amdgcn_wmma_f32_16x16x32_f16(false, (a), false, (b), (short)0, (c), false, false)

__global__ __launch_bounds__(256)
void lstm_head_kernel(const float* __restrict__ x,
                      const float* __restrict__ Wtp,  const float* __restrict__ btp,
                      const float* __restrict__ Wih0, const float* __restrict__ Whh0,
                      const float* __restrict__ bih0, const float* __restrict__ bhh0,
                      const float* __restrict__ Wih1, const float* __restrict__ Whh1,
                      const float* __restrict__ bih1, const float* __restrict__ bhh1,
                      const float* __restrict__ Wfc1, const float* __restrict__ bfc1,
                      const float* __restrict__ Wfc2, const float* __restrict__ bfc2,
                      float* __restrict__ out, int n, int ntiles)
{
    // Block-shared f16 copies of the LSTM recurrent/input weight matrices.
    // Whh0, Wih1, Whh1 are all [128][32] row-major; B[k, col] = W[col, k], so a
    // B-fragment (16 contiguous K per lane) is 16 contiguous halfs of one row.
    __shared__ __align__(32) _Float16 whh0_s[128 * 32];
    __shared__ __align__(32) _Float16 wih1_s[128 * 32];
    __shared__ __align__(32) _Float16 whh1_s[128 * 32];
    // Per-wave scratch: h-state transpose buffers (A-matrix source) + x slice.
    __shared__ __align__(16) _Float16 h0_s[NWAVES][16 * PITCH];
    __shared__ __align__(16) _Float16 h1_s[NWAVES][16 * PITCH];
    __shared__ float s_s[NWAVES][16 * TSTEPS];

    const int tid = threadIdx.x;
    for (int i = tid; i < 128 * 32; i += 256) {
        whh0_s[i] = (_Float16)Whh0[i];
        wih1_s[i] = (_Float16)Wih1[i];
        whh1_s[i] = (_Float16)Whh1[i];
    }
    __syncthreads();

    const int wv   = tid >> 5;
    const int lane = tid & 31;
    const int tile = blockIdx.x * NWAVES + wv;
    if (tile >= ntiles) return;          // wave-uniform: EXEC stays all-ones below
    const int base = tile << 4;

    const int nA    = lane & 15;               // B/C/D column, A row (M)
    const int hiOff = (lane < 16) ? 0 : 8;     // C/D row offset: element r -> node r+hiOff
    const int kb    = (lane < 16) ? 0 : 16;    // B-matrix K base for this lane
    const int k0    = (lane < 16) ? 0 : 8;     // unused: loadA reads contiguous K kb..kb+15

    _Float16* h0w = h0_s[wv];
    _Float16* h1w = h1_s[wv];
    float*    sw  = s_s[wv];

    // Stage this tile's 12 time-step scalars (x columns 52..63) into LDS.
    if (lane < 16) {
        int node = base + lane;
        if (node >= n) node = n - 1;
        const float* row = x + (size_t)node * 64 + 52;
        __builtin_prefetch(row, 0, 0);
        #pragma unroll
        for (int i = 0; i < TSTEPS; ++i) sw[lane * TSTEPS + i] = row[i];
    }
    // Zero-init hidden-state transpose buffers (h0 = h1 = 0).
    for (int i = lane; i < 16 * PITCH; i += 32) {
        h0w[i] = (_Float16)0.0f;
        h1w[i] = (_Float16)0.0f;
    }

    // Fold the per-step input projection: xt = s*Wtp + btp  =>
    // xt @ Wih0^T = s*u + v.  Precompute per-lane for the 8 gate columns.
    float u8[8], w08[8], b18[8];
    #pragma unroll
    for (int j = 0; j < 8; ++j) {
        const int col = j * 16 + nA;
        float su = 0.0f, sv = 0.0f;
        #pragma unroll
        for (int k = 0; k < 16; ++k) {
            const float wv2 = Wih0[col * 16 + k];
            su = fmaf(Wtp[k], wv2, su);
            sv = fmaf(btp[k], wv2, sv);
        }
        u8[j]  = su;
        w08[j] = sv + bih0[col] + bhh0[col];
        b18[j] = bih1[col] + bhh1[col];
    }

    v8f c0[2], c1[2];
    #pragma unroll
    for (int r = 0; r < 8; ++r) { c0[0][r] = 0.f; c0[1][r] = 0.f; c1[0][r] = 0.f; c1[1][r] = 0.f; }

    // A-fragment loader (16-bit 16x32 A layout): lane holds row M=nA,
    // halfs 0..15 = K kb..kb+15 (contiguous, matching B).
    auto loadA = [&](const _Float16* hb) -> v16h {
        const v8h lo = *(const v8h*)(hb + nA * PITCH + kb);
        const v8h hi = *(const v8h*)(hb + nA * PITCH + kb + 8);
        return __builtin_shufflevector(lo, hi, 0, 1, 2, 3, 4, 5, 6, 7,
                                               8, 9, 10, 11, 12, 13, 14, 15);
    };
    // B-fragment loader (16-bit 32x16 B layout): 16 contiguous K at column j*16+nA.
    auto loadB = [&](const _Float16* wb, int j) -> v16h {
        return *(const v16h*)(wb + (j * 16 + nA) * 32 + kb);
    };

    #pragma unroll 1
    for (int t = 0; t < TSTEPS; ++t) {
        // Keep the 24 weight B-fragments in LDS (re-read each step) instead of
        // letting the compiler hoist 192 VGPRs of weights across the loop:
        // staying under 256 VGPRs avoids s_set_vgpr_msb mode churn and keeps
        // occupancy up. LDS bandwidth for 48 ds_load_b128/step is negligible.
        asm volatile("" ::: "memory");

        // ---- layer 0: gates = s*u + bias + h0 @ Whh0^T ----
        const v16h a0 = loadA(h0w);
        float sv8[8];
        #pragma unroll
        for (int r = 0; r < 8; ++r) sv8[r] = sw[(r + hiOff) * TSTEPS + t];

        v8f g[8];
        #pragma unroll
        for (int j = 0; j < 8; ++j) {
            v8f acc;
            #pragma unroll
            for (int r = 0; r < 8; ++r) acc[r] = fmaf(sv8[r], u8[j], w08[j]);
            g[j] = WMMA16(a0, loadB(whh0_s, j), acc);
        }
        // elementwise cell update (torch gate order i,f,g,o = cols 0..31,32..63,64..95,96..127)
        #pragma unroll
        for (int jt = 0; jt < 2; ++jt) {
            #pragma unroll
            for (int r = 0; r < 8; ++r) {
                const float iv = sigm_f(g[jt][r]);
                const float fv = sigm_f(g[2 + jt][r]);
                const float gv = tanh_f(g[4 + jt][r]);
                const float ov = sigm_f(g[6 + jt][r]);
                const float cv = fmaf(fv, c0[jt][r], iv * gv);
                c0[jt][r] = cv;
                const float hv = ov * tanh_f(cv);
                h0w[(r + hiOff) * PITCH + jt * 16 + nA] = (_Float16)hv;
            }
        }

        // ---- layer 1: gates = bias + h0_new @ Wih1^T + h1 @ Whh1^T ----
        const v16h a0n = loadA(h0w);
        const v16h a1  = loadA(h1w);
        #pragma unroll
        for (int j = 0; j < 8; ++j) {
            v8f acc;
            #pragma unroll
            for (int r = 0; r < 8; ++r) acc[r] = b18[j];
            acc  = WMMA16(a0n, loadB(wih1_s, j), acc);
            g[j] = WMMA16(a1,  loadB(whh1_s, j), acc);
        }
        #pragma unroll
        for (int jt = 0; jt < 2; ++jt) {
            #pragma unroll
            for (int r = 0; r < 8; ++r) {
                const float iv = sigm_f(g[jt][r]);
                const float fv = sigm_f(g[2 + jt][r]);
                const float gv = tanh_f(g[4 + jt][r]);
                const float ov = sigm_f(g[6 + jt][r]);
                const float cv = fmaf(fv, c1[jt][r], iv * gv);
                c1[jt][r] = cv;
                const float hv = ov * tanh_f(cv);
                h1w[(r + hiOff) * PITCH + jt * 16 + nA] = (_Float16)hv;
            }
        }
    }

    // ---- FC head: out = relu(h1 @ Wfc1 + bfc1) @ Wfc2 + bfc2 ----
    const v16h a1f = loadA(h1w);
    v16h bf;
    #pragma unroll
    for (int e = 0; e < 16; ++e) bf[e] = (_Float16)Wfc1[(kb + e) * 16 + nA];
    v8f d;
    const float bn = bfc1[nA];
    #pragma unroll
    for (int r = 0; r < 8; ++r) d[r] = bn;
    d = WMMA16(a1f, bf, d);

    const float w2 = Wfc2[nA];
    const float b2 = bfc2[0];
    #pragma unroll
    for (int r = 0; r < 8; ++r) {
        float v = fmaxf(d[r], 0.0f) * w2;
        v += __shfl_xor(v, 1);
        v += __shfl_xor(v, 2);
        v += __shfl_xor(v, 4);
        v += __shfl_xor(v, 8);
        const int node = base + r + hiOff;
        if (nA == 0 && node < n) out[node] = v + b2;
    }
}

extern "C" void kernel_launch(void* const* d_in, const int* in_sizes, int n_in,
                              void* d_out, int out_size, void* d_ws, size_t ws_size,
                              hipStream_t stream) {
    (void)n_in; (void)out_size; (void)d_ws; (void)ws_size;
    const float* x    = (const float*)d_in[0];
    // d_in[1..9]: edge_index + GCN weights -> dead in the reference output, skipped.
    const float* Wtp  = (const float*)d_in[10];
    const float* btp  = (const float*)d_in[11];
    const float* Wih0 = (const float*)d_in[12];
    const float* Whh0 = (const float*)d_in[13];
    const float* bih0 = (const float*)d_in[14];
    const float* bhh0 = (const float*)d_in[15];
    const float* Wih1 = (const float*)d_in[16];
    const float* Whh1 = (const float*)d_in[17];
    const float* bih1 = (const float*)d_in[18];
    const float* bhh1 = (const float*)d_in[19];
    const float* Wfc1 = (const float*)d_in[20];
    const float* bfc1 = (const float*)d_in[21];
    const float* Wfc2 = (const float*)d_in[22];
    const float* bfc2 = (const float*)d_in[23];
    float* out = (float*)d_out;

    const int n      = in_sizes[0] / 64;        // N (F = 64)
    const int ntiles = (n + 15) / 16;           // 16-node tiles, one per wave
    const int blocks = (ntiles + NWAVES - 1) / NWAVES;

    lstm_head_kernel<<<blocks, 256, 0, stream>>>(
        x, Wtp, btp, Wih0, Whh0, bih0, bhh0,
        Wih1, Whh1, bih1, bhh1, Wfc1, bfc1, Wfc2, bfc2,
        out, n, ntiles);
}
